// AFM_32607391711646
// MI455X (gfx1250) — matrix operand order repaired and checked by reading, not compiled, not run on hardware
//
#include <hip/hip_runtime.h>

typedef __attribute__((ext_vector_type(16))) __bf16 v16bf;
typedef __attribute__((ext_vector_type(8)))  float  v8f;

union FragAB { v16bf v; uint4 u[2]; };

#define NCH 64
#define MCH 64
#define KK  9
#define HH  128
#define WW  128

#define XS_ELEMS   (3 * 130 * 72)          // bf16, 56160 B
#define WF_ELEMS   (MCH * NCH * KK)        // 36864 bf16, 73728 B
#define XS_BYTES   (XS_ELEMS * 2)
#define WF_BYTES   (WF_ELEMS * 2)
#define SMEM_BYTES (XS_BYTES + WF_BYTES)   // 129888 B dynamic LDS

// ---------------- Kernel 1: per-(b,c) global average of x ----------------
__global__ __launch_bounds__(256)
void k_gmean(const float* __restrict__ x, float* __restrict__ g) {
  const int blk = blockIdx.x;                 // b*64 + c
  const float* xp = x + (size_t)blk * (HH * WW);
  __shared__ float red[256];
  float s = 0.f;
  for (int i = threadIdx.x; i < HH * WW; i += 256) s += xp[i];
  red[threadIdx.x] = s;
  __syncthreads();
  for (int off = 128; off > 0; off >>= 1) {
    if (threadIdx.x < off) red[threadIdx.x] += red[threadIdx.x + off];
    __syncthreads();
  }
  if (threadIdx.x == 0) g[blk] = red[0] * (1.f / (HH * WW));
}

// ---------------- Kernel 2: attention MLPs -> ratio[b,9], bias[b,64] ----------------
__global__ __launch_bounds__(64)
void k_attn(const float* __restrict__ g,
            const float* __restrict__ a2w1, const float* __restrict__ a2b1,
            const float* __restrict__ a2w2, const float* __restrict__ a2b2,
            const float* __restrict__ a3w1, const float* __restrict__ a3b1,
            const float* __restrict__ a3w2, const float* __restrict__ a3b2,
            float* __restrict__ ratio, float* __restrict__ bias) {
  const int b = blockIdx.x;
  const int t = threadIdx.x;                  // 64 threads
  __shared__ float gs[64], h2[16], h3[64];
  gs[t] = g[b * 64 + t];
  __syncthreads();
  if (t < KK) {
    float s = a2b1[t];
    for (int c = 0; c < 64; ++c) s += gs[c] * a2w1[t * 64 + c];
    h2[t] = s > 0.f ? s : 0.f;
  }
  {
    float s = a3b1[t];
    for (int c = 0; c < 64; ++c) s += gs[c] * a3w1[t * 64 + c];
    h3[t] = s > 0.f ? s : 0.f;
  }
  __syncthreads();
  if (t < KK) {
    float s = a2b2[t];
    for (int j = 0; j < KK; ++j) s += h2[j] * a2w2[t * KK + j];
    ratio[b * KK + t] = s;
  }
  {
    float s = a3b2[t];
    for (int j = 0; j < 64; ++j) s += h3[j] * a3w2[t * 64 + j];
    bias[b * 64 + t] = s;
  }
}

// ------- Kernel 3: pack conv weight into bf16, A-fragment sliced lane order -------
// fragment f = ((t*2 + cb)*4 + mt)
// element e = f*512 + u*256 + lane*8 + j2   (u = which 16B half of the lane's 16 vals)
// A 16x32 bf16 layout: lanes 0-15: K = {0..7, 16..23}; lanes 16-31: K = {8..15, 24..31}
__global__ __launch_bounds__(256)
void k_packw(const float* __restrict__ w, __bf16* __restrict__ wfrag) {
  const int e    = blockIdx.x * 256 + threadIdx.x;   // < 36864
  const int f    = e >> 9;
  const int r    = e & 511;
  const int u    = r >> 8;                           // 16B half
  const int lane = (r >> 3) & 31;
  const int j    = u * 8 + (r & 7);                  // 0..15 within lane fragment
  const int mt   = f & 3;
  const int cb   = (f >> 2) & 1;
  const int t    = f >> 3;
  const int kl   = j + ((j >= 8) ? 8 : 0) + ((lane >= 16) ? 8 : 0);
  const int c    = cb * 32 + kl;
  const int M    = mt * 16 + (lane & 15);
  wfrag[e] = (__bf16)w[(M * NCH + c) * KK + t];
}

// ---------------- Kernel 4: main WMMA conv-attention GEMM ----------------
__global__ __launch_bounds__(256)
void k_main(const float* __restrict__ x, const float* __restrict__ atw1,
            const __bf16* __restrict__ wfrag, const float* __restrict__ ratio,
            const float* __restrict__ bias, float* __restrict__ out) {
  const int ph   = blockIdx.x;
  const int b    = blockIdx.y;
  const int tid  = threadIdx.x;
  const int wave = tid >> 5;
  const int lane = tid & 31;

  extern __shared__ __align__(16) char smem[];
  __bf16* xs = (__bf16*)smem;                 // [3][130][72] x tile (bf16)
  __bf16* wl = (__bf16*)(smem + XS_BYTES);    // packed weight fragments (bf16)
  __shared__ float ratio_s[KK];
  __shared__ float bias_s[64];

  if (tid < KK) ratio_s[tid] = ratio[b * KK + tid];
  if (tid < 64) bias_s[tid]  = bias[b * 64 + tid];

  // Stage packed weights: straight coalesced memcpy (layout identical to global).
  {
    const uint4* src = (const uint4*)wfrag;
    uint4* dst = (uint4*)wl;
    for (int i = tid; i < WF_BYTES / 16; i += 256) dst[i] = src[i];
  }

  // Stage x rows ph-1..ph+1: coalesced global row reads, transposed LDS scatter.
  for (int s = wave; s < NCH * 3; s += 8) {
    const int c = s / 3, r = s - c * 3;
    const int h = ph + r - 1;
    const bool hv = (h >= 0) && (h < HH);
    const float* xrow = x + (((size_t)(b * NCH + c)) * HH + (hv ? h : 0)) * WW;
    for (int col = lane; col < 130; col += 32) {
      const int w = col - 1;
      const float v = (hv && w >= 0 && w < WW) ? xrow[w] : 0.f;
      xs[(r * 130 + col) * 72 + c] = (__bf16)v;
    }
  }
  __syncthreads();

  const int col16 = lane & 15;
  const int pw    = wave * 16 + col16;          // this lane's output column
  const int khalf = (lane >= 16) ? 8 : 0;       // B-fragment K half per lane group
  const int alane = lane * 8;                   // A-fragment per-lane slice (bf16 units)

  const v8f vzero = {0.f, 0.f, 0.f, 0.f, 0.f, 0.f, 0.f, 0.f};
  v8f acc[4] = {vzero, vzero, vzero, vzero};

  for (int t = 0; t < KK; ++t) {
    const int dh = t / 3 - 1;
    const int dw = t - (t / 3) * 3 - 1;
    const float a_val =
        atw1[(((size_t)(b * KK + t)) * HH + ph) * WW + pw] * ratio_s[t];

    v8f part[4] = {vzero, vzero, vzero, vzero};
#pragma unroll
    for (int cb = 0; cb < 2; ++cb) {
      FragAB B;
      const uint4* bp = (const uint4*)(xs + ((dh + 1) * 130 + (pw + dw + 1)) * 72
                                          + cb * 32 + khalf);
      B.u[0] = bp[0];
      B.u[1] = bp[2];
#pragma unroll
      for (int mt = 0; mt < 4; ++mt) {
        FragAB A;
        const __bf16* ab = wl + ((t * 2 + cb) * 4 + mt) * 512 + alane;
        A.u[0] = *(const uint4*)(ab);
        A.u[1] = *(const uint4*)(ab + 256);
        part[mt] = __builtin_amdgcn_wmma_f32_16x16x32_bf16(
            false, A.v, false, B.v, (short)0, part[mt], false, false);
      }
    }
#pragma unroll
    for (int mt = 0; mt < 4; ++mt) acc[mt] += a_val * part[mt];
  }

  // Epilogue: + bias, store. C/D layout: VGPR i holds M = i (+8 for lanes>=16).
  const int rbase = (lane < 16) ? 0 : 8;
#pragma unroll
  for (int mt = 0; mt < 4; ++mt) {
#pragma unroll
    for (int i = 0; i < 8; ++i) {
      const int mm = mt * 16 + rbase + i;
      out[(((size_t)(b * MCH + mm)) * HH + ph) * WW + pw] = acc[mt][i] + bias_s[mm];
    }
  }
}

// ---------------- host launcher ----------------
extern "C" void kernel_launch(void* const* d_in, const int* in_sizes, int n_in,
                              void* d_out, int out_size, void* d_ws, size_t ws_size,
                              hipStream_t stream) {
  const float* x    = (const float*)d_in[0];
  const float* atw1 = (const float*)d_in[1];
  const float* wgt  = (const float*)d_in[2];
  const float* a2w1 = (const float*)d_in[3];
  const float* a2b1 = (const float*)d_in[4];
  const float* a2w2 = (const float*)d_in[5];
  const float* a2b2 = (const float*)d_in[6];
  const float* a3w1 = (const float*)d_in[7];
  const float* a3b1 = (const float*)d_in[8];
  const float* a3w2 = (const float*)d_in[9];
  const float* a3b2 = (const float*)d_in[10];
  float* out = (float*)d_out;

  char* ws = (char*)d_ws;
  float*  g     = (float*)(ws);           // 512 floats
  float*  ratio = (float*)(ws + 2048);    // 72 floats
  float*  bias  = (float*)(ws + 2560);    // 512 floats
  __bf16* wfrag = (__bf16*)(ws + 8192);   // 36864 bf16 (72 KB)

  k_gmean<<<dim3(8 * NCH), dim3(256), 0, stream>>>(x, g);
  k_attn<<<dim3(8), dim3(64), 0, stream>>>(g, a2w1, a2b1, a2w2, a2b2,
                                           a3w1, a3b1, a3w2, a3b2, ratio, bias);
  k_packw<<<dim3(WF_ELEMS / 256), dim3(256), 0, stream>>>(wgt, wfrag);
  k_main<<<dim3(HH, 8), dim3(256), SMEM_BYTES, stream>>>(x, atw1, wfrag, ratio,
                                                         bias, out);
}
